// GMS_45810121179782
// MI455X (gfx1250) — matrix operand order repaired and checked by reading, not compiled, not run on hardware
//
#include <hip/hip_runtime.h>

typedef __bf16 bf16;
typedef __attribute__((ext_vector_type(16))) __bf16 v16bf;
typedef __attribute__((ext_vector_type(8)))  float   v8f;

static const int NVARS  = 50000;
static const int NLITS  = 100000;   // 2*NVARS
static const int NCLS   = 210000;
static const int ROUNDS = 8;
#define DIM 128

// ---------------- device helpers ----------------
__device__ __forceinline__ unsigned short f2bf_bits(float f) {
  unsigned u = __builtin_bit_cast(unsigned, f);
  u += 0x7fffu + ((u >> 16) & 1u);           // round-to-nearest-even
  return (unsigned short)(u >> 16);
}
__device__ __forceinline__ bf16 f2bf(float f) {
  unsigned short s = f2bf_bits(f);
  return __builtin_bit_cast(bf16, s);
}
__device__ __forceinline__ float sigm(float x) { return 1.0f / (1.0f + __expf(-x)); }

// A-matrix fragment (16x32 bf16, row-major source). ISA 7.12.2:
// lanes 0-15: V0-3 hold K = 8g..8g+7, V4-7 hold K = 16+8g..16+8g+7 (g = lane>>4)
__device__ __forceinline__ v16bf ldA(const bf16* row, int k0, int g) {
  union { v16bf v; uint4 u[2]; } r;
  r.u[0] = *(const uint4*)(row + k0 + 8 * g);
  r.u[1] = *(const uint4*)(row + k0 + 16 + 8 * g);
  return r.v;
}
// Same A fragment but sourced from f32 (message buffers), packed to bf16 on the fly
__device__ __forceinline__ v16bf ldA_f32(const float* row, int k0, int g) {
  union { v16bf v; unsigned short s[16]; } r;
  const float4 x0 = *(const float4*)(row + k0 + 8 * g);
  const float4 x1 = *(const float4*)(row + k0 + 8 * g + 4);
  const float4 x2 = *(const float4*)(row + k0 + 16 + 8 * g);
  const float4 x3 = *(const float4*)(row + k0 + 16 + 8 * g + 4);
  r.s[0] = f2bf_bits(x0.x); r.s[1] = f2bf_bits(x0.y); r.s[2] = f2bf_bits(x0.z); r.s[3] = f2bf_bits(x0.w);
  r.s[4] = f2bf_bits(x1.x); r.s[5] = f2bf_bits(x1.y); r.s[6] = f2bf_bits(x1.z); r.s[7] = f2bf_bits(x1.w);
  r.s[8]  = f2bf_bits(x2.x); r.s[9]  = f2bf_bits(x2.y); r.s[10] = f2bf_bits(x2.z); r.s[11] = f2bf_bits(x2.w);
  r.s[12] = f2bf_bits(x3.x); r.s[13] = f2bf_bits(x3.y); r.s[14] = f2bf_bits(x3.z); r.s[15] = f2bf_bits(x3.w);
  return r.v;
}
// B-matrix fragment (32x16 = W^T chunk): column n of B = row n of W (row-major, ld=K).
// same K layout as A: V0-3 hold K = 8g..8g+7, V4-7 hold K = 16+8g..16+8g+7 (g = lane>>4).
__device__ __forceinline__ v16bf ldB(const bf16* wrow, int k0, int g) {
  union { v16bf v; uint4 u[2]; } r;
  const bf16* p = wrow + k0 + 8 * g;
  r.u[0] = *(const uint4*)(p);
  r.u[1] = *(const uint4*)(p + 16);
  return r.v;
}
__device__ __forceinline__ v8f wmma_bf16(v16bf a, v16bf b, v8f c) {
  return __builtin_amdgcn_wmma_f32_16x16x32_bf16(false, a, false, b, (short)0, c, false, false);
}

// ---------------- weight prep (runs once per launch, negligible) ----------------
// Fuse gat: Wg = Wo @ Wv  (128x128), bg = Wo @ bv + bo
__global__ __launch_bounds__(256) void prep_gat(
    const float* __restrict__ mha_in_w, const float* __restrict__ mha_in_b,
    const float* __restrict__ mha_out_w, const float* __restrict__ mha_out_b,
    bf16* __restrict__ Wg, float* __restrict__ bg) {
  const int h = blockIdx.x;
  const float* Wv = mha_in_w + (size_t)h * 384 * DIM + 256 * DIM;  // rows 2d..3d
  const float* bv = mha_in_b + h * 384 + 256;
  const float* Wo = mha_out_w + (size_t)h * DIM * DIM;
  const float* bo = mha_out_b + h * DIM;
  for (int o = threadIdx.x; o < DIM * DIM; o += blockDim.x) {
    int m = o >> 7, k = o & 127;
    float s = 0.f;
    for (int j = 0; j < DIM; ++j) s += Wo[m * DIM + j] * Wv[j * DIM + k];
    Wg[(size_t)h * DIM * DIM + o] = f2bf(s);
  }
  for (int m = threadIdx.x; m < DIM; m += blockDim.x) {
    float s = bo[m];
    for (int j = 0; j < DIM; ++j) s += Wo[m * DIM + j] * bv[j];
    bg[h * DIM + m] = s;
  }
}

// Concatenate [wih | whh] -> bf16 [512 x (Kx+128)], bias = bih + bhh
__global__ __launch_bounds__(256) void prep_lstm(
    const float* __restrict__ wih, const float* __restrict__ whh,
    const float* __restrict__ bih, const float* __restrict__ bhh,
    bf16* __restrict__ Wcat, float* __restrict__ bias, int Kx) {
  const int Kcat = Kx + DIM;
  const int total = 512 * Kcat;
  const int stride = gridDim.x * blockDim.x;
  for (int o = blockIdx.x * blockDim.x + threadIdx.x; o < total; o += stride) {
    int n = o / Kcat, k = o - n * Kcat;
    float v = (k < Kx) ? wih[(size_t)n * Kx + k] : whh[(size_t)n * DIM + (k - Kx)];
    Wcat[o] = f2bf(v);
  }
  for (int n = blockIdx.x * blockDim.x + threadIdx.x; n < 512; n += stride)
    bias[n] = bih[n] + bhh[n];
}

__global__ __launch_bounds__(256) void cvt_bf16(const float* __restrict__ s,
                                                bf16* __restrict__ d, int n) {
  int i = blockIdx.x * blockDim.x + threadIdx.x;
  if (i < n) d[i] = f2bf(s[i]);
}

// h0 = broadcast(w[:,0] + b)
__global__ __launch_bounds__(256) void init_state(const float* __restrict__ w,
                                                  const float* __restrict__ b,
                                                  bf16* __restrict__ H, int n) {
  int i = blockIdx.x * blockDim.x + threadIdx.x;
  if (i < n * DIM) { int d = i & 127; H[i] = f2bf(w[d] + b[d]); }
}

// ---------------- fused gat GEMM: G = H @ Wg^T + bg ----------------
// One wave = 32 rows (2 M-tiles) x 128 cols: each B fragment feeds two wmmas,
// halving exposed load latency and doubling intensity vs the weight stream.
__global__ __launch_bounds__(32) void gat_gemm(const bf16* __restrict__ H,
                                               const bf16* __restrict__ Wg,
                                               const float* __restrict__ bg,
                                               float* __restrict__ G, int nrows) {
  const int r0base = blockIdx.x * 32;
  const bool two = (r0base + 16) < nrows;          // block-uniform tail guard
  const int lane = threadIdx.x, l16 = lane & 15, g = lane >> 4;
  const bf16* arow0 = H + (size_t)(r0base + l16) * DIM;
  const bf16* arow1 = H + (size_t)(r0base + (two ? 16 : 0) + l16) * DIM;
  const bf16* wrow = Wg + (size_t)l16 * DIM;
  v8f acc0[8] = {}, acc1[8] = {};
#pragma unroll
  for (int ks = 0; ks < 4; ++ks) {
    const int k0 = ks * 32;
    v16bf a0 = ldA(arow0, k0, g);
    v16bf a1 = ldA(arow1, k0, g);
#pragma unroll
    for (int nt = 0; nt < 8; ++nt) {
      v16bf b = ldB(wrow + (size_t)nt * 16 * DIM, k0, g);
      acc0[nt] = wmma_bf16(a0, b, acc0[nt]);
      acc1[nt] = wmma_bf16(a1, b, acc1[nt]);
    }
  }
#pragma unroll
  for (int nt = 0; nt < 8; ++nt) {
    const int n = nt * 16 + l16;
    const float bv = bg[n];
#pragma unroll
    for (int j = 0; j < 8; ++j) {
      const int m = j + 8 * g;
      G[(size_t)(r0base + m) * DIM + n] = acc0[nt][j] + bv;
      if (two) G[(size_t)(r0base + 16 + m) * DIM + n] = acc1[nt][j] + bv;
    }
  }
}

// ---------------- edge scatter: M[dst] += G[src]  (f32 atomics into L2-resident buffer) ----------------
__global__ __launch_bounds__(256) void scatter_add(const float* __restrict__ G,
                                                   const int* __restrict__ src,
                                                   const int* __restrict__ dst,
                                                   float* __restrict__ M, int nE) {
  const int t = blockIdx.x * 256 + threadIdx.x;
  const int e = t >> 7, c = t & 127;
  if (e >= nE) return;
  const int s = src[e], d = dst[e];
  atomicAdd(&M[(size_t)d * DIM + c], G[(size_t)s * DIM + c]);
}

// ---------------- fused LSTM: gates GEMM (WMMA) + cell update ----------------
// Block = 4 waves (one gate each) x 32 rows (2 M-tiles).
// MODE 0 (clauses, KCAT=256):  k<128: X(f32)          ; k>=128: Hb[r]
// MODE 1 (lits,    KCAT=384):  k<128: X(f32)          ; 128..255: Hb[flip(r)] ; k>=256: Hb[r]
template <int KCAT, int MODE>
__global__ __launch_bounds__(128) void lstm_fused(
    const float* __restrict__ X, const bf16* __restrict__ Hb,
    const bf16* __restrict__ Wcat, const float* __restrict__ bias,
    float* __restrict__ Cc, bf16* __restrict__ Hout, int nrows, int nv) {
  __shared__ __align__(16) float gbuf[32 * 512];   // 64 KB: gates for 32 rows
  const int r0base = blockIdx.x * 32;
  const bool two = (r0base + 16) < nrows;          // block-uniform tail guard
  const int wave = threadIdx.x >> 5;               // gate index (i,f,g,o)
  const int lane = threadIdx.x & 31, l16 = lane & 15, g = lane >> 4;
  const int r0 = r0base + l16;
  const int r1 = r0base + (two ? 16 : 0) + l16;    // clamped when tail
  const bf16* wbase = Wcat + (size_t)(wave * 128 + l16) * KCAT;

  v8f acc0[8] = {}, acc1[8] = {};
#pragma unroll
  for (int ks = 0; ks < KCAT / 32; ++ks) {
    const int k0 = ks * 32;
    v16bf a0, a1;
    if (k0 < 128) {
      a0 = ldA_f32(X + (size_t)r0 * DIM, k0, g);
      a1 = ldA_f32(X + (size_t)r1 * DIM, k0, g);
    } else if (MODE == 0) {
      a0 = ldA(Hb + (size_t)r0 * DIM, k0 - 128, g);
      a1 = ldA(Hb + (size_t)r1 * DIM, k0 - 128, g);
    } else if (k0 < 256) {
      const int rf0 = (r0 < nv) ? r0 + nv : r0 - nv;   // flipped literal
      const int rf1 = (r1 < nv) ? r1 + nv : r1 - nv;
      a0 = ldA(Hb + (size_t)rf0 * DIM, k0 - 128, g);
      a1 = ldA(Hb + (size_t)rf1 * DIM, k0 - 128, g);
    } else {
      a0 = ldA(Hb + (size_t)r0 * DIM, k0 - 256, g);
      a1 = ldA(Hb + (size_t)r1 * DIM, k0 - 256, g);
    }
#pragma unroll
    for (int nt = 0; nt < 8; ++nt) {
      v16bf b = ldB(wbase + (size_t)nt * 16 * KCAT, k0, g);
      acc0[nt] = wmma_bf16(a0, b, acc0[nt]);
      acc1[nt] = wmma_bf16(a1, b, acc1[nt]);
    }
  }
#pragma unroll
  for (int nt = 0; nt < 8; ++nt) {
    const int n = wave * 128 + nt * 16 + l16;
    const float bv = bias[n];
#pragma unroll
    for (int j = 0; j < 8; ++j) {
      gbuf[(j + 8 * g) * 512 + n] = acc0[nt][j] + bv;
      gbuf[(16 + j + 8 * g) * 512 + n] = acc1[nt][j] + bv;
    }
  }
  __syncthreads();

  const int c = threadIdx.x;                       // 0..127, coalesced over columns
  const int mmax = two ? 32 : 16;
#pragma unroll 8
  for (int m = 0; m < mmax; ++m) {
    const float iv = gbuf[m * 512 + c];
    const float fv = gbuf[m * 512 + 128 + c];
    const float gv = gbuf[m * 512 + 256 + c];
    const float ov = gbuf[m * 512 + 384 + c];
    const size_t idx = (size_t)(r0base + m) * DIM + c;
    const float cold = Cc[idx];
    const float cn = sigm(fv) * cold + sigm(iv) * tanhf(gv);
    Cc[idx] = cn;
    Hout[idx] = f2bf(sigm(ov) * tanhf(cn));
  }
}

// ---------------- fused 3-layer MLP over vars (one wave = 16 rows; runs once) ----------------
__global__ __launch_bounds__(32) void mlp_fused(
    const bf16* __restrict__ Lh, const bf16* __restrict__ W1, const float* __restrict__ b1,
    const bf16* __restrict__ W2, const float* __restrict__ b2,
    const float* __restrict__ w3, const float* __restrict__ b3,
    float* __restrict__ out, int nv) {
  __shared__ __align__(16) bf16  h1s[16 * DIM];
  __shared__ __align__(16) float h2s[16 * DIM];
  const int tile = blockIdx.x;
  const int lane = threadIdx.x, l16 = lane & 15, g = lane >> 4;
  const int r = tile * 16 + l16;

  // GEMM1: x = [Lh[v] | Lh[v+nv]] (K=256) @ W1^T
  v8f acc[8] = {};
#pragma unroll
  for (int ks = 0; ks < 8; ++ks) {
    const int k0 = ks * 32;
    v16bf a = (k0 < 128) ? ldA(Lh + (size_t)r * DIM, k0, g)
                         : ldA(Lh + (size_t)(r + nv) * DIM, k0 - 128, g);
#pragma unroll
    for (int nt = 0; nt < 8; ++nt) {
      v16bf b = ldB(W1 + (size_t)(nt * 16 + l16) * 256, k0, g);
      acc[nt] = wmma_bf16(a, b, acc[nt]);
    }
  }
#pragma unroll
  for (int nt = 0; nt < 8; ++nt) {
    const int n = nt * 16 + l16;
    const float bv = b1[n];
#pragma unroll
    for (int j = 0; j < 8; ++j) {
      float v = acc[nt][j] + bv;
      h1s[(j + 8 * g) * DIM + n] = f2bf(v > 0.f ? v : 0.f);
    }
  }
  __syncthreads();

  // GEMM2: K=128, A from LDS
  v8f acc2[8] = {};
#pragma unroll
  for (int ks = 0; ks < 4; ++ks) {
    const int k0 = ks * 32;
    v16bf a = ldA(&h1s[l16 * DIM], k0, g);
#pragma unroll
    for (int nt = 0; nt < 8; ++nt) {
      v16bf b = ldB(W2 + (size_t)(nt * 16 + l16) * DIM, k0, g);
      acc2[nt] = wmma_bf16(a, b, acc2[nt]);
    }
  }
#pragma unroll
  for (int nt = 0; nt < 8; ++nt) {
    const int n = nt * 16 + l16;
    const float bv = b2[n];
#pragma unroll
    for (int j = 0; j < 8; ++j) {
      float v = acc2[nt][j] + bv;
      h2s[(j + 8 * g) * DIM + n] = v > 0.f ? v : 0.f;
    }
  }
  __syncthreads();

  // vote = h2 @ w3^T + b3
  if (lane < 16) {
    float s = b3[0];
    for (int cc = 0; cc < DIM; ++cc) s += h2s[lane * DIM + cc] * w3[cc];
    out[tile * 16 + lane] = s;
  }
}

// ---------------- host orchestration ----------------
extern "C" void kernel_launch(void* const* d_in, const int* in_sizes, int n_in,
                              void* d_out, int out_size, void* d_ws, size_t ws_size,
                              hipStream_t stream) {
  const float* l_init_w  = (const float*)d_in[0];
  const float* l_init_b  = (const float*)d_in[1];
  const float* c_init_w  = (const float*)d_in[2];
  const float* c_init_b  = (const float*)d_in[3];
  const float* mha_in_w  = (const float*)d_in[4];
  const float* mha_in_b  = (const float*)d_in[5];
  const float* mha_out_w = (const float*)d_in[6];
  const float* mha_out_b = (const float*)d_in[7];
  const float* L_wih = (const float*)d_in[8];
  const float* L_whh = (const float*)d_in[9];
  const float* L_bih = (const float*)d_in[10];
  const float* L_bhh = (const float*)d_in[11];
  const float* C_wih = (const float*)d_in[12];
  const float* C_whh = (const float*)d_in[13];
  const float* C_bih = (const float*)d_in[14];
  const float* C_bhh = (const float*)d_in[15];
  const float* mlp_w1 = (const float*)d_in[16];
  const float* mlp_b1 = (const float*)d_in[17];
  const float* mlp_w2 = (const float*)d_in[18];
  const float* mlp_b2 = (const float*)d_in[19];
  const float* mlp_w3 = (const float*)d_in[20];
  const float* mlp_b3 = (const float*)d_in[21];
  const int* e_src[3] = {(const int*)d_in[22], (const int*)d_in[24], (const int*)d_in[26]}; // lit
  const int* e_dst[3] = {(const int*)d_in[23], (const int*)d_in[25], (const int*)d_in[27]}; // clause
  const int  e_cnt[3] = {in_sizes[22], in_sizes[24], in_sizes[26]};

  // bump allocator over workspace
  char* base = (char*)d_ws;
  size_t off = 0;
  auto alloc = [&](size_t bytes) -> void* {
    void* r = base + off;
    off += (bytes + 255) & ~(size_t)255;
    return r;
  };
  bf16*  LhA = (bf16*)alloc((size_t)NLITS * DIM * 2);
  bf16*  LhB = (bf16*)alloc((size_t)NLITS * DIM * 2);
  float* Lc  = (float*)alloc((size_t)NLITS * DIM * 4);
  bf16*  ChA = (bf16*)alloc((size_t)NCLS * DIM * 2);
  bf16*  ChB = (bf16*)alloc((size_t)NCLS * DIM * 2);
  float* Cc  = (float*)alloc((size_t)NCLS * DIM * 4);
  float* LCm = (float*)alloc((size_t)NCLS * DIM * 4);
  float* CLm = (float*)alloc((size_t)NLITS * DIM * 4);
  float* G   = (float*)alloc((size_t)NCLS * DIM * 4);     // gat output, reused both dirs
  bf16*  Wg  = (bf16*)alloc((size_t)6 * DIM * DIM * 2);
  float* bg  = (float*)alloc((size_t)6 * DIM * 4);
  bf16*  WC  = (bf16*)alloc((size_t)512 * 256 * 2);
  float* bC  = (float*)alloc(512 * 4);
  bf16*  WL  = (bf16*)alloc((size_t)512 * 384 * 2);
  float* bL  = (float*)alloc(512 * 4);
  bf16*  W1b = (bf16*)alloc((size_t)DIM * 256 * 2);
  bf16*  W2b = (bf16*)alloc((size_t)DIM * DIM * 2);

  // weight prep + state init
  prep_gat<<<6, 256, 0, stream>>>(mha_in_w, mha_in_b, mha_out_w, mha_out_b, Wg, bg);
  prep_lstm<<<(512 * 256 + 255) / 256, 256, 0, stream>>>(C_wih, C_whh, C_bih, C_bhh, WC, bC, 128);
  prep_lstm<<<(512 * 384 + 255) / 256, 256, 0, stream>>>(L_wih, L_whh, L_bih, L_bhh, WL, bL, 256);
  cvt_bf16<<<(DIM * 256 + 255) / 256, 256, 0, stream>>>(mlp_w1, W1b, DIM * 256);
  cvt_bf16<<<(DIM * DIM + 255) / 256, 256, 0, stream>>>(mlp_w2, W2b, DIM * DIM);
  init_state<<<(NLITS * DIM + 255) / 256, 256, 0, stream>>>(l_init_w, l_init_b, LhA, NLITS);
  init_state<<<(NCLS * DIM + 255) / 256, 256, 0, stream>>>(c_init_w, c_init_b, ChA, NCLS);
  hipMemsetAsync(Lc, 0, (size_t)NLITS * DIM * 4, stream);
  hipMemsetAsync(Cc, 0, (size_t)NCLS * DIM * 4, stream);

  const int gridL = (NLITS + 31) / 32;   // 3125
  const int gridC = (NCLS + 31) / 32;    // 6563 (last block: 16 rows)

  bf16 *Lh = LhA, *Lhn = LhB, *Ch = ChA, *Chn = ChB;
  for (int rd = 0; rd < ROUNDS; ++rd) {
    // ---- L -> C ----
    hipMemsetAsync(LCm, 0, (size_t)NCLS * DIM * 4, stream);
    for (int h = 0; h < 3; ++h) {
      gat_gemm<<<gridL, 32, 0, stream>>>(Lh, Wg + (size_t)h * DIM * DIM, bg + h * DIM, G, NLITS);
      scatter_add<<<(e_cnt[h] * DIM + 255) / 256, 256, 0, stream>>>(G, e_src[h], e_dst[h], LCm, e_cnt[h]);
    }
    lstm_fused<256, 0><<<gridC, 128, 0, stream>>>(LCm, Ch, WC, bC, Cc, Chn, NCLS, 0);
    { bf16* t = Ch; Ch = Chn; Chn = t; }

    // ---- C -> L ----
    hipMemsetAsync(CLm, 0, (size_t)NLITS * DIM * 4, stream);
    for (int h = 0; h < 3; ++h) {
      gat_gemm<<<gridC, 32, 0, stream>>>(Ch, Wg + (size_t)(h + 3) * DIM * DIM, bg + (h + 3) * DIM, G, NCLS);
      scatter_add<<<(e_cnt[h] * DIM + 255) / 256, 256, 0, stream>>>(G, e_dst[h], e_src[h], CLm, e_cnt[h]);
    }
    lstm_fused<384, 1><<<gridL, 128, 0, stream>>>(CLm, Lh, WL, bL, Lc, Lhn, NLITS, NVARS);
    { bf16* t = Lh; Lh = Lhn; Lhn = t; }
  }

  // ---- final MLP vote ----
  mlp_fused<<<NVARS / 16, 32, 0, stream>>>(Lh, W1b, mlp_b1, W2b, mlp_b2, mlp_w3, mlp_b3,
                                           (float*)d_out, NVARS);
  (void)n_in; (void)out_size; (void)ws_size; (void)in_sizes;
}
